// FlashGemma3Attention_83811991814214
// MI455X (gfx1250) — matrix operand order repaired and checked by reading, not compiled, not run on hardware
//
#include <hip/hip_runtime.h>
#include <hip/hip_bf16.h>
#include <stdint.h>

// ---------------- problem constants ----------------
#define B_      2
#define S_      2048
#define HID_    2560
#define H_      8
#define KV_     4
#define D_      256
#define WINDOW_ 1024
#define T_      (B_ * S_)             // 4096 tokens
#define NQKV_   ((H_ + 2 * KV_) * D_) // 4096
#define SCALE_  0.0625f               // 256^-0.5
#define EPS_    1e-6f

typedef __bf16 bf16;
typedef __attribute__((address_space(3))) bf16 lds_bf16;
typedef __attribute__((ext_vector_type(16))) __bf16 v16bf;
typedef __attribute__((ext_vector_type(8)))  __bf16 v8bf;
typedef __attribute__((ext_vector_type(8)))  float  v8f;

__device__ __forceinline__ v8f zero8() {
  v8f z = {0.f, 0.f, 0.f, 0.f, 0.f, 0.f, 0.f, 0.f};
  return z;
}

// D = A(16x32 bf16) * B(32x16 bf16) + C(16x16 f32), f32 accumulate
__device__ __forceinline__ v8f wmma_bf16(v16bf a, v16bf b, v8f c) {
  return __builtin_amdgcn_wmma_f32_16x16x32_bf16(
      /*neg_a=*/false, a, /*neg_b=*/false, b,
      /*c_mod=*/(short)0, c, /*reuse_a=*/false, /*reuse_b=*/false);
}

// Build a 16x32 bf16 A-fragment (contiguous k order, same as the B-fragment loads): per lane (row = lane&15, half = lane>>4):
//   elems 0..7  <- K = klo + 0..7   (klo = base + half*16)
//   elems 8..15 <- K = khi + 0..7   (khi = base + half*16 + 8)
__device__ __forceinline__ v16bf build_afrag(const bf16* rowp, int klo, int khi) {
  v8bf lo = *(const v8bf*)(rowp + klo);
  v8bf hi = *(const v8bf*)(rowp + khi);
  v16bf r;
#pragma unroll
  for (int e = 0; e < 8; ++e) { r[e] = lo[e]; r[e + 8] = hi[e]; }
  return r;
}

// gfx1250 async copy global -> LDS (16B per lane), tracked by ASYNCcnt.
// VDST slot carries the LDS byte address, VADDR the 64-bit global address (GV mode).
// The LDS operand is ptrtoint of a real addrspace(3) pointer: this captures the
// shared array's address so the optimizer cannot fold its loads to undef.
__device__ __forceinline__ void async_load_b128(const bf16* g, lds_bf16* l) {
  unsigned long long ga = (unsigned long long)(uintptr_t)g;
  uint32_t lo = (uint32_t)(uintptr_t)l;  // LDS byte offset
  asm volatile("global_load_async_to_lds_b128 %0, %1, off"
               :: "v"(lo), "v"(ga)
               : "memory");
}
__device__ __forceinline__ void wait_async(int pending) {
  if (pending)
    asm volatile("s_wait_asynccnt 0x6" ::: "memory");  // 1 stage (6 ops) in flight
  else
    asm volatile("s_wait_asynccnt 0x0" ::: "memory");
}

// ---------------- f32 -> bf16 convert ----------------
__global__ void k_cvt_bf16(const float* __restrict__ src, bf16* __restrict__ dst, int n) {
  int i = blockIdx.x * blockDim.x + threadIdx.x;
  int stride = gridDim.x * blockDim.x;
  for (; i < n; i += stride) dst[i] = (bf16)src[i];
}

// ---------------- block-tiled WMMA GEMM: C[MxN] = A[MxK] * B[NxK]^T ----------------
// Block (8 waves) computes a 128x256 tile; each wave a 64x64 sub-tile (4x4 WMMA
// fragments, 128 accumulator VGPRs). Per 32-wide K-chunk, the A(128x32) and
// B(256x32) slices are staged into LDS with global_load_async_to_lds_b128
// (ASYNCcnt), double-buffered so the copy of chunk k+1 hides behind the 16
// WMMAs of chunk k.
__global__ void __launch_bounds__(256)
k_gemm_bf16(const bf16* __restrict__ A, const bf16* __restrict__ B,
            float* __restrict__ C, int M, int N, int K) {
  __shared__ alignas(128) bf16 smem[2][(128 + 256) * 32];  // 2 x 24KB

  const int tid = threadIdx.x;
  const int lane = tid & 31, wave = tid >> 5;
  const int nblk = N >> 8;  // block tiles along N (256 wide)
  const int tm0 = (blockIdx.x / nblk) << 7;
  const int tn0 = (blockIdx.x % nblk) << 8;
  const int mw = (wave >> 2) << 6;  // wave sub-tile offsets in block tile
  const int nw = (wave & 3) << 6;
  const int r = lane & 15, half = lane >> 4;

  // per-thread staging: A tile = 512 x 16B chunks (2/thread), B = 1024 (4/thread)
  auto stage = [&](int k0, int buf) {
    bf16* As = &smem[buf][0];
    bf16* Bs = &smem[buf][128 * 32];
#pragma unroll
    for (int i = 0; i < 2; ++i) {
      int idx = tid + 256 * i;
      int row = idx >> 2, cc = idx & 3;  // 4 x 16B chunks per 64B row
      async_load_b128(A + (size_t)(tm0 + row) * K + k0 + cc * 8,
                      (lds_bf16*)(As + row * 32 + cc * 8));
    }
#pragma unroll
    for (int i = 0; i < 4; ++i) {
      int idx = tid + 256 * i;
      int row = idx >> 2, cc = idx & 3;
      async_load_b128(B + (size_t)(tn0 + row) * K + k0 + cc * 8,
                      (lds_bf16*)(Bs + row * 32 + cc * 8));
    }
  };

  v8f acc[4][4];
#pragma unroll
  for (int mt = 0; mt < 4; ++mt)
#pragma unroll
    for (int nt = 0; nt < 4; ++nt) acc[mt][nt] = zero8();

  const int nit = K >> 5;
  stage(0, 0);
  for (int it = 0; it < nit; ++it) {
    const int buf = it & 1;
    const int have_next = (it + 1 < nit);
    if (have_next) stage((it + 1) << 5, buf ^ 1);  // overlap copy with compute
    wait_async(have_next);  // this chunk's 6 ops done; next stage may be in flight
    __syncthreads();        // every wave's staged data visible

    const bf16* As = &smem[buf][0];
    const bf16* Bs = &smem[buf][128 * 32];
    v16bf af[4], bfr[4];
#pragma unroll
    for (int mt = 0; mt < 4; ++mt)
      af[mt] = build_afrag(As + (mw + mt * 16 + r) * 32, half * 16, half * 16 + 8);
#pragma unroll
    for (int nt = 0; nt < 4; ++nt)
      bfr[nt] = *(const v16bf*)(Bs + (nw + nt * 16 + r) * 32 + half * 16);
#pragma unroll
    for (int mt = 0; mt < 4; ++mt)
#pragma unroll
      for (int nt = 0; nt < 4; ++nt)
        acc[mt][nt] = wmma_bf16(af[mt], bfr[nt], acc[mt][nt]);

    __syncthreads();  // all waves done reading buf before it is restaged
  }

  // C layout: VGPR q -> M = q + 8*half, N = lane&15
#pragma unroll
  for (int mt = 0; mt < 4; ++mt)
#pragma unroll
    for (int nt = 0; nt < 4; ++nt) {
      float* cp = C + (size_t)(tm0 + mw + mt * 16 + half * 8) * N +
                  (tn0 + nw + nt * 16 + r);
#pragma unroll
      for (int q = 0; q < 8; ++q) cp[(size_t)q * N] = acc[mt][nt][q];
    }
}

// ---------------- fused RMSNorm + RoPE for q and k heads ----------------
__global__ void __launch_bounds__(128)
k_norm_rope(const float* __restrict__ qkv, const float* __restrict__ cosT,
            const float* __restrict__ sinT, const float* __restrict__ qw,
            const float* __restrict__ kw, bf16* __restrict__ qb,
            bf16* __restrict__ kb) {
  const int t = blockIdx.x;
  const int wave = threadIdx.x >> 5, lane = threadIdx.x & 31;
  for (int u = wave; u < H_ + KV_; u += 4) {
    const float* src;
    const float* w;
    bf16* dst;
    if (u < H_) {
      src = qkv + (size_t)t * NQKV_ + u * D_;
      w = qw;
      dst = qb + ((size_t)t * H_ + u) * D_;
    } else {
      int j = u - H_;
      src = qkv + (size_t)t * NQKV_ + (H_ + j) * D_;
      w = kw;
      dst = kb + ((size_t)t * KV_ + j) * D_;
    }
    float v[8];
    float ss = 0.f;
#pragma unroll
    for (int i = 0; i < 8; ++i) {
      v[i] = src[lane + 32 * i];
      ss += v[i] * v[i];
    }
#pragma unroll
    for (int mk = 16; mk >= 1; mk >>= 1) ss += __shfl_xor(ss, mk, 32);
    float rstd = rsqrtf(ss * (1.0f / D_) + EPS_);
    float nrm[8];
#pragma unroll
    for (int i = 0; i < 8; ++i) nrm[i] = v[i] * rstd * w[lane + 32 * i];
#pragma unroll
    for (int i = 0; i < 4; ++i) {
      float c = cosT[(size_t)t * (D_ / 2) + lane + 32 * i];
      float s = sinT[(size_t)t * (D_ / 2) + lane + 32 * i];
      float x1 = nrm[i], x2 = nrm[i + 4];
      dst[lane + 32 * i]       = (bf16)(x1 * c - x2 * s);
      dst[lane + 32 * i + 128] = (bf16)(x2 * c + x1 * s);
    }
  }
}

// ---------------- V transpose: qkv f32 [t][3072 + j*256 + d] -> bf16 [b][kv][d][s] ----------------
__global__ void k_v_transpose(const float* __restrict__ qkv, bf16* __restrict__ vt,
                              int total) {
  int i = blockIdx.x * blockDim.x + threadIdx.x;
  int stride = gridDim.x * blockDim.x;
  for (; i < total; i += stride) {
    int s = i & (S_ - 1);
    int d = (i >> 11) & (D_ - 1);
    int j = (i >> 19) & (KV_ - 1);
    int b = i >> 21;
    vt[i] = (bf16)qkv[(size_t)(b * S_ + s) * NQKV_ + (H_ + KV_) * D_ + j * D_ + d];
  }
}

// ---------------- flash attention, sliding window 1024, WMMA QK^T and P*V ----------------
__global__ void __launch_bounds__(256)
k_attn(const bf16* __restrict__ qb, const bf16* __restrict__ kb,
       const bf16* __restrict__ vt, bf16* __restrict__ ob) {
  __shared__ alignas(64) bf16 Ps[8][16][32];  // 8 waves x (16 queries x 32 keys)
  const int lane = threadIdx.x & 31, wave = threadIdx.x >> 5;
  const int gw = blockIdx.x * 8 + wave;
  const int bh = gw >> 7;     // 16 (b,h) pairs
  const int qblk = gw & 127;  // 128 query blocks of 16
  const int b = bh >> 3, h = bh & 7, kvh = h >> 1;
  const int q0 = qblk << 4;
  const int r = lane & 15, half = lane >> 4, col = lane & 15;

  const bf16* qrow = qb + (((size_t)(b * S_ + q0 + r)) * H_ + h) * D_;

  float mrow[8], lrow[8];
  v8f acc[16];
#pragma unroll
  for (int i = 0; i < 8; ++i) { mrow[i] = -3.0e38f; lrow[i] = 0.f; }
#pragma unroll
  for (int i = 0; i < 16; ++i) acc[i] = zero8();

  int kt0 = q0 - (WINDOW_ - 1);
  if (kt0 < 0) kt0 = 0;
  kt0 &= ~31;

  for (int kt = kt0; kt < q0 + 16; kt += 32) {
    // ---- S = Q * K^T for keys [kt, kt+32): two 16-key halves ----
    v8f s0 = zero8(), s1 = zero8();
    const bf16* k0p = kb + (((size_t)(b * S_ + kt + col)) * KV_ + kvh) * D_;
    const bf16* k1p = kb + (((size_t)(b * S_ + kt + 16 + col)) * KV_ + kvh) * D_;
    __builtin_prefetch(k0p + 32 * KV_ * D_, 0, 3);  // next key tile's K rows
#pragma unroll
    for (int dch = 0; dch < 8; ++dch) {
      v16bf qf = build_afrag(qrow, dch * 32 + half * 16, dch * 32 + half * 16 + 8);
      v16bf kf0 = *(const v16bf*)(k0p + dch * 32 + half * 16);
      s0 = wmma_bf16(qf, kf0, s0);
      v16bf kf1 = *(const v16bf*)(k1p + dch * 32 + half * 16);
      s1 = wmma_bf16(qf, kf1, s1);
    }
    // ---- online softmax update (rows live in one 16-lane half) ----
#pragma unroll
    for (int rr = 0; rr < 8; ++rr) {
      int q = q0 + rr + 8 * half;
      int key0 = kt + col, key1 = kt + 16 + col;
      float a0 = (key0 <= q && (q - key0) < WINDOW_) ? s0[rr] * SCALE_ : -1e30f;
      float a1 = (key1 <= q && (q - key1) < WINDOW_) ? s1[rr] * SCALE_ : -1e30f;
      float mx = fmaxf(a0, a1);
#pragma unroll
      for (int mk = 8; mk >= 1; mk >>= 1) mx = fmaxf(mx, __shfl_xor(mx, mk, 32));
      float mnew = fmaxf(mrow[rr], mx);
      float scl = __expf(mrow[rr] - mnew);
      float p0 = __expf(a0 - mnew), p1 = __expf(a1 - mnew);
      float ps = p0 + p1;
#pragma unroll
      for (int mk = 8; mk >= 1; mk >>= 1) ps += __shfl_xor(ps, mk, 32);
      lrow[rr] = lrow[rr] * scl + ps;
      mrow[rr] = mnew;
#pragma unroll
      for (int n = 0; n < 16; ++n) acc[n][rr] *= scl;
      Ps[wave][rr + 8 * half][col]      = (bf16)p0;
      Ps[wave][rr + 8 * half][col + 16] = (bf16)p1;
    }
    // ---- O += P * V : A-frag of P from LDS, B-frag of V contiguous (d-major) ----
    const bf16* prow = &Ps[wave][r][0];
    v16bf pf = build_afrag(prow, half * 16, half * 16 + 8);
    const bf16* vbase =
        vt + ((size_t)(b * KV_ + kvh)) * D_ * S_ + (size_t)col * S_ + kt + half * 16;
#pragma unroll
    for (int nch = 0; nch < 16; ++nch) {
      v16bf vf = *(const v16bf*)(vbase + (size_t)(nch * 16) * S_);
      acc[nch] = wmma_bf16(pf, vf, acc[nch]);
    }
  }
  // ---- epilogue: normalize and store bf16 [t][h][d] for the output GEMM ----
#pragma unroll
  for (int rr = 0; rr < 8; ++rr) {
    float inv = 1.0f / lrow[rr];
    bf16* op = ob + (((size_t)(b * S_ + q0 + rr + 8 * half)) * H_ + h) * D_ + col;
#pragma unroll
    for (int nch = 0; nch < 16; ++nch) op[nch * 16] = (bf16)(acc[nch][rr] * inv);
  }
}

// ---------------- host-side orchestration ----------------
extern "C" void kernel_launch(void* const* d_in, const int* in_sizes, int n_in,
                              void* d_out, int out_size, void* d_ws, size_t ws_size,
                              hipStream_t stream) {
  (void)in_sizes; (void)n_in; (void)out_size; (void)ws_size;
  const float* hidden = (const float*)d_in[0];
  const float* cosT   = (const float*)d_in[1];
  const float* sinT   = (const float*)d_in[2];
  const float* w_qkv  = (const float*)d_in[3];
  const float* q_norm = (const float*)d_in[4];
  const float* k_norm = (const float*)d_in[5];
  const float* w_o    = (const float*)d_in[6];
  float* out = (float*)d_out;

  char* ws = (char*)d_ws;
  size_t off = 0;
  auto alloc = [&](size_t bytes) -> void* {
    off = (off + 255) & ~(size_t)255;
    void* p = ws + off;
    off += bytes;
    return p;
  };
  bf16*  hid_bf  = (bf16*)alloc((size_t)T_ * HID_ * 2);
  bf16*  wqkv_bf = (bf16*)alloc((size_t)NQKV_ * HID_ * 2);
  bf16*  wo_bf   = (bf16*)alloc((size_t)HID_ * (H_ * D_) * 2);
  float* qkv     = (float*)alloc((size_t)T_ * NQKV_ * 4);
  bf16*  q_bf    = (bf16*)alloc((size_t)T_ * H_ * D_ * 2);
  bf16*  k_bf    = (bf16*)alloc((size_t)T_ * KV_ * D_ * 2);
  bf16*  vt_bf   = (bf16*)alloc((size_t)B_ * KV_ * D_ * S_ * 2);
  bf16*  attn_bf = (bf16*)alloc((size_t)T_ * H_ * D_ * 2);

  // 1. f32 -> bf16 casts (memory-bound, trivial vs. 23.3 TB/s)
  k_cvt_bf16<<<2048, 256, 0, stream>>>(hidden, hid_bf, T_ * HID_);
  k_cvt_bf16<<<2048, 256, 0, stream>>>(w_qkv, wqkv_bf, NQKV_ * HID_);
  k_cvt_bf16<<<2048, 256, 0, stream>>>(w_o, wo_bf, HID_ * H_ * D_);

  // 2. QKV projection: (4096x2560) * (4096x2560)^T -> f32 qkv
  k_gemm_bf16<<<(T_ / 128) * (NQKV_ / 256), 256, 0, stream>>>(
      hid_bf, wqkv_bf, qkv, T_, NQKV_, HID_);

  // 3. RMSNorm + RoPE on q/k heads -> bf16
  k_norm_rope<<<T_, 128, 0, stream>>>(qkv, cosT, sinT, q_norm, k_norm, q_bf, k_bf);

  // 4. V -> d-major bf16 layout
  k_v_transpose<<<2048, 256, 0, stream>>>(qkv, vt_bf, B_ * KV_ * D_ * S_);

  // 5. sliding-window flash attention
  k_attn<<<(B_ * H_ * (S_ / 16)) / 8, 256, 0, stream>>>(q_bf, k_bf, vt_bf, attn_bf);

  // 6. output projection: (4096x2048) * (2560x2048)^T -> f32 out
  k_gemm_bf16<<<(T_ / 128) * (HID_ / 256), 256, 0, stream>>>(
      attn_bf, wo_bf, out, T_, HID_, H_ * D_);
}
